// TransformerLayer_4080218931572
// MI455X (gfx1250) — matrix operand order repaired and checked by reading, not compiled, not run on hardware
//
#include <hip/hip_runtime.h>
#include <hip/hip_bf16.h>

typedef __bf16 bf16;
typedef __attribute__((ext_vector_type(16))) __bf16      v16bf;
typedef __attribute__((ext_vector_type(8)))  float       v8f;
typedef __attribute__((ext_vector_type(8)))  unsigned int v8u;
typedef __attribute__((ext_vector_type(4)))  int         i32x4;

#define WMMA_BF16(a, b, c) \
  __builtin_amdgcn_wmma_f32_16x16x32_bf16(false, (a), false, (b), (short)0, (c), false, false)

// --- CDNA5 async global->LDS path (ASYNCcnt), guarded so either toolchain compiles
#if __has_builtin(__builtin_amdgcn_global_load_async_to_lds_b128) && \
    __has_builtin(__builtin_amdgcn_s_wait_asynccnt)
#define HAVE_ASYNC_LDS 1
// builtin prototype (from hipcc diagnostic): (int4 AS1*, int4 AS3*, Imm offset, Imm cpol)
typedef __attribute__((address_space(1))) i32x4 gi32x4;
typedef __attribute__((address_space(3))) i32x4 li32x4;
#endif

static __device__ inline void async_cp16(const bf16* g, bf16* l) {
#ifdef HAVE_ASYNC_LDS
  void* gv = (void*)(const void*)g;  // strip const, keep generic
  void* lv = (void*)l;
  __builtin_amdgcn_global_load_async_to_lds_b128((gi32x4*)gv, (li32x4*)lv, 0, 0);
#else
  *reinterpret_cast<uint4*>(l) = *reinterpret_cast<const uint4*>(g);
#endif
}
static __device__ inline void async_wait() {
#ifdef HAVE_ASYNC_LDS
  __builtin_amdgcn_s_wait_asynccnt(0);
#endif
}

// ---------------------------------------------------------------------------
// Fragment builders (this file's contiguous k order k = 16*hi + s, not the ISA table order; wave32)
// A-matrix 16x32 bf16: lane(0..15)=row M; lanes 0-15 hold K 0..15,
// lanes 16-31 hold K 16..31; VGPR v holds K pair (kbase, kbase+1).
static __device__ inline v16bf load_frag_a(const bf16* __restrict__ rowp, int hi) {
  union { v8u u; v16bf b; } f;
#pragma unroll
  for (int v = 0; v < 8; ++v) {
    int kbase = 2 * v + 16 * hi;
    f.u[v] = *reinterpret_cast<const unsigned int*>(rowp + kbase);
  }
  return f.b;
}
// B-matrix 32x16 bf16: lane(0..15)=col N; lane half selects K 0-15 / 16-31.
// colp must point at (col, 16*hi) in a [N][K] (k-contiguous) LDS tile.
static __device__ inline v16bf load_frag_b(const bf16* __restrict__ colp) {
  union { v8u u; v16bf b; } f;
#pragma unroll
  for (int v = 0; v < 8; ++v)
    f.u[v] = *reinterpret_cast<const unsigned int*>(colp + 2 * v);
  return f.b;
}

// ---------------------------------------------------------------------------
// fp32 -> bf16 conversion
__global__ __launch_bounds__(256) void cvt_f32_bf16(const float* __restrict__ in,
                                                    bf16* __restrict__ out, int n) {
  int i = blockIdx.x * 256 + threadIdx.x;
  if (i < n) out[i] = (bf16)in[i];
}

// ---------------------------------------------------------------------------
// Tiled bf16 WMMA GEMM: C[M,N] = act(A[M,K] @ B[K,N]); A,B bf16 row-major.
// 128x128 block tile, 8 waves, each wave 4x2 of 16x16 WMMA tiles, K step 32.
// Double-buffered LDS; A tile staged with async global->LDS when available.
template <bool RELU, bool OUT_BF16>
__global__ __launch_bounds__(256)
void gemm_wmma_bf16(const bf16* __restrict__ A, const bf16* __restrict__ B,
                    float* __restrict__ Cf, bf16* __restrict__ Cb,
                    int M, int N, int K) {
  __shared__ bf16 As[2][128][32];  // [m][k]
  __shared__ bf16 Bs[2][128][32];  // [n][k]  (B tile transposed: k-contiguous)

  const int tid = threadIdx.x;
  const int lane = tid & 31, wave = tid >> 5;
  const int lo = lane & 15, hi = lane >> 4;
  const int bm = blockIdx.y * 128, bn = blockIdx.x * 128;
  const int wm = (wave & 1) * 64;   // 4 m-tiles of 16
  const int wn = (wave >> 1) * 32;  // 2 n-tiles of 16

  v8f acc[4][2];
#pragma unroll
  for (int i = 0; i < 4; ++i)
#pragma unroll
    for (int j = 0; j < 2; ++j) { v8f z = {}; acc[i][j] = z; }

  const int arow = tid >> 1, akoff = (tid & 1) * 16;
  const int bk = tid >> 3, bn0 = (tid & 7) * 16;

  auto stage = [&](int buf, int kt) {
    const bf16* asrc = &A[(size_t)(bm + arow) * K + kt + akoff];
    async_cp16(asrc,     &As[buf][arow][akoff]);
    async_cp16(asrc + 8, &As[buf][arow][akoff + 8]);
    const bf16* bsrc = &B[(size_t)(kt + bk) * N + bn + bn0];
#pragma unroll
    for (int i = 0; i < 16; ++i) Bs[buf][bn0 + i][bk] = bsrc[i];
  };

  stage(0, 0);
  async_wait();
  __syncthreads();

  const int nk = K >> 5;
  for (int t = 0; t < nk; ++t) {
    const int cur = t & 1;
    if (t + 1 < nk) stage(cur ^ 1, (t + 1) << 5);  // overlap next tile's traffic

    // gather all fragments first, then issue the 8 WMMAs back-to-back
    v16bf bfrag[2], afrag[4];
#pragma unroll
    for (int nt = 0; nt < 2; ++nt)
      bfrag[nt] = load_frag_b(&Bs[cur][wn + nt * 16 + lo][16 * hi]);
#pragma unroll
    for (int mt = 0; mt < 4; ++mt)
      afrag[mt] = load_frag_a(&As[cur][wm + mt * 16 + lo][0], hi);
#pragma unroll
    for (int mt = 0; mt < 4; ++mt)
#pragma unroll
      for (int nt = 0; nt < 2; ++nt)
        acc[mt][nt] = WMMA_BF16(afrag[mt], bfrag[nt], acc[mt][nt]);

    async_wait();
    __syncthreads();
  }

  // epilogue: C/D layout = col lo per lane, rows r + 8*hi per VGPR
#pragma unroll
  for (int mt = 0; mt < 4; ++mt)
#pragma unroll
    for (int nt = 0; nt < 2; ++nt)
#pragma unroll
      for (int r = 0; r < 8; ++r) {
        int row = bm + wm + mt * 16 + r + 8 * hi;
        int col = bn + wn + nt * 16 + lo;
        float v = acc[mt][nt][r];
        if constexpr (RELU) v = fmaxf(v, 0.0f);
        if constexpr (OUT_BF16) Cb[(size_t)row * N + col] = (bf16)v;
        else                    Cf[(size_t)row * N + col] = v;
      }
}

// ---------------------------------------------------------------------------
// Flash attention: one block per (128 q rows, head, batch); 8 waves, each wave
// owns a 16-row Q tile. K/V/pad tiles (32 keys) staged in LDS, online softmax,
// scores and P@V on bf16 WMMA. V = x (no Wv in the reference module).
__global__ __launch_bounds__(256)
void flash_attn(const bf16* __restrict__ Q, const bf16* __restrict__ Km,
                const bf16* __restrict__ V, const unsigned char* __restrict__ pad,
                float* __restrict__ Out, int S, int H, int DH) {
  __shared__ bf16 Ks[32][128];      // [key][d]   (d-contiguous -> Kt B-frags)
  __shared__ bf16 Vt[128][32];      // [d][key]   (key-contiguous -> V B-frags)
  __shared__ bf16 Ps[8][16][32];    // per-wave P transpose buffer
  __shared__ unsigned char pads[32];

  const int b = blockIdx.z, h = blockIdx.y;
  const int D = H * DH;
  const int qb = blockIdx.x * 128;
  const int tid = threadIdx.x, lane = tid & 31, wave = tid >> 5;
  const int lo = lane & 15, hi = lane >> 4;
  const int qrow = qb + wave * 16;
  const float scale = 0.08838834764831845f;  // 1/sqrt(128)

  // Q fragments: 4 chunks of K=32 covering d_h = 128 (held in registers)
  v16bf qf[4];
  {
    const bf16* qptr = Q + (size_t)(b * S + qrow + lo) * D + h * DH;
#pragma unroll
    for (int c = 0; c < 4; ++c) qf[c] = load_frag_a(qptr + c * 32, hi);
  }

  v8f o[8];
  float mrun[8], lrun[8];
#pragma unroll
  for (int i = 0; i < 8; ++i) { v8f z = {}; o[i] = z; mrun[i] = -1e9f; lrun[i] = 0.0f; }

  const int qmax = qrow + 15;
  const int kr = tid >> 3, d0 = (tid & 7) * 16;
  for (int jt = 0; jt < qb + 128; jt += 32) {  // uniform trip count across block
    // stage K tile (async), transposed V tile, pad bytes
    {
      const bf16* ksrc = Km + (size_t)(b * S + jt + kr) * D + h * DH + d0;
      async_cp16(ksrc,     &Ks[kr][d0]);
      async_cp16(ksrc + 8, &Ks[kr][d0 + 8]);
      const bf16* vs = V + (size_t)(b * S + jt + kr) * D + h * DH + d0;
#pragma unroll
      for (int i = 0; i < 16; ++i) Vt[d0 + i][kr] = vs[i];
      if (tid < 32) pads[tid] = pad[(size_t)b * S + jt + tid];
    }
    async_wait();
    __syncthreads();

    if (jt <= qmax) {
      // scores: S16x32 = Q(16x128) @ K^T — load all 8 frags, then 8 WMMAs
      v16bf kf[8];
#pragma unroll
      for (int c = 0; c < 4; ++c)
#pragma unroll
        for (int nt = 0; nt < 2; ++nt)
          kf[c * 2 + nt] = load_frag_b(&Ks[nt * 16 + lo][c * 32 + 16 * hi]);
      v8f s[2];
      { v8f z = {}; s[0] = z; s[1] = z; }
#pragma unroll
      for (int c = 0; c < 4; ++c)
#pragma unroll
        for (int nt = 0; nt < 2; ++nt)
          s[nt] = WMMA_BF16(qf[c], kf[c * 2 + nt], s[nt]);

      // scale + causal/pad mask (col j = lane, rows r+8*hi in VGPRs)
#pragma unroll
      for (int nt = 0; nt < 2; ++nt) {
        int j = jt + nt * 16 + lo;
        bool pv = pads[nt * 16 + lo] != 0;
#pragma unroll
        for (int r = 0; r < 8; ++r) {
          int i = qrow + r + 8 * hi;
          s[nt][r] = (pv && j <= i) ? s[nt][r] * scale : -1e9f;
        }
      }
      // online softmax: row reductions across the 16 lanes of each half-wave
      float pr0[8], pr1[8];
#pragma unroll
      for (int r = 0; r < 8; ++r) {
        float mx = fmaxf(s[0][r], s[1][r]);
#pragma unroll
        for (int off = 8; off >= 1; off >>= 1) mx = fmaxf(mx, __shfl_xor(mx, off, 16));
        float mnew = fmaxf(mrun[r], mx);
        float corr = __expf(mrun[r] - mnew);
        float p0 = __expf(s[0][r] - mnew);
        float p1 = __expf(s[1][r] - mnew);
        float rs = p0 + p1;
#pragma unroll
        for (int off = 8; off >= 1; off >>= 1) rs += __shfl_xor(rs, off, 16);
        lrun[r] = lrun[r] * corr + rs;
        mrun[r] = mnew;
        pr0[r] = p0; pr1[r] = p1;
#pragma unroll
        for (int nt = 0; nt < 8; ++nt) o[nt][r] *= corr;
      }
      // C-layout -> A-layout transpose of P through per-wave LDS (bf16)
#pragma unroll
      for (int r = 0; r < 8; ++r) {
        Ps[wave][r + 8 * hi][lo]      = (bf16)pr0[r];
        Ps[wave][r + 8 * hi][16 + lo] = (bf16)pr1[r];
      }
      v16bf pf = load_frag_a(&Ps[wave][lo][0], hi);  // same-wave DS ops in-order
      // O(16x128) += P(16x32) @ V(32x128): frag-batched WMMA issue
#pragma unroll
      for (int g = 0; g < 2; ++g) {
        v16bf vf[4];
#pragma unroll
        for (int q = 0; q < 4; ++q)
          vf[q] = load_frag_b(&Vt[(g * 4 + q) * 16 + lo][16 * hi]);
#pragma unroll
        for (int q = 0; q < 4; ++q)
          o[g * 4 + q] = WMMA_BF16(pf, vf[q], o[g * 4 + q]);
      }
    }
    __syncthreads();
  }

#pragma unroll
  for (int nt = 0; nt < 8; ++nt)
#pragma unroll
    for (int r = 0; r < 8; ++r) {
      int i = qrow + r + 8 * hi;
      Out[(size_t)(b * S + i) * D + h * DH + nt * 16 + lo] = o[nt][r] / lrun[r];
    }
}

// ---------------------------------------------------------------------------
// Residual add + LayerNorm; optional bf16 copy for the next GEMM's A operand.
template <bool BF16OUT>
__global__ __launch_bounds__(256)
void resid_ln(const float* __restrict__ a, const float* __restrict__ r,
              const float* __restrict__ g, const float* __restrict__ be,
              float* __restrict__ out, bf16* __restrict__ outb, int D) {
  __shared__ float red[256];
  const int row = blockIdx.x, t = threadIdx.x;
  const float* pa = a + (size_t)row * D;
  const float* pr = r + (size_t)row * D;

  float s = 0.0f;
  for (int i = t; i < D; i += 256) s += pa[i] + pr[i];
  red[t] = s; __syncthreads();
  for (int o = 128; o > 0; o >>= 1) { if (t < o) red[t] += red[t + o]; __syncthreads(); }
  float mu = red[0] / D;
  __syncthreads();

  float v = 0.0f;
  for (int i = t; i < D; i += 256) { float d = pa[i] + pr[i] - mu; v += d * d; }
  red[t] = v; __syncthreads();
  for (int o = 128; o > 0; o >>= 1) { if (t < o) red[t] += red[t + o]; __syncthreads(); }
  float inv = rsqrtf(red[0] / D + 1e-5f);

  for (int i = t; i < D; i += 256) {
    float y = (pa[i] + pr[i] - mu) * inv * g[i] + be[i];
    out[(size_t)row * D + i] = y;
    if constexpr (BF16OUT) outb[(size_t)row * D + i] = (bf16)y;
  }
}

// ---------------------------------------------------------------------------
extern "C" void kernel_launch(void* const* d_in, const int* in_sizes, int n_in,
                              void* d_out, int out_size, void* d_ws, size_t ws_size,
                              hipStream_t stream) {
  (void)in_sizes; (void)n_in; (void)out_size; (void)ws_size;
  constexpr int B = 4, S = 1024, D = 2048, DFF = 8192, H = 16, DH = 128;
  constexpr int M = B * S;                // 4096 rows
  constexpr size_t BSD = (size_t)B * S * D;

  const float* x  = (const float*)d_in[0];
  const float* Wq = (const float*)d_in[1];
  const float* Wk = (const float*)d_in[2];
  const float* W1 = (const float*)d_in[3];
  const float* W2 = (const float*)d_in[4];
  const float* g1 = (const float*)d_in[5];
  const float* b1 = (const float*)d_in[6];
  const float* g2 = (const float*)d_in[7];
  const float* b2 = (const float*)d_in[8];
  /* d_in[9] = causal_mask (handled analytically) */
  const unsigned char* pad = (const unsigned char*)d_in[10];
  float* out = (float*)d_out;

  char* ws = (char*)d_ws;
  size_t off = 0;
  auto alloc = [&](size_t bytes) { char* p = ws + off; off = (off + bytes + 255) & ~(size_t)255; return p; };
  bf16*  xb   = (bf16*)alloc(BSD * 2);
  bf16*  Wqb  = (bf16*)alloc((size_t)D * D * 2);
  bf16*  Wkb  = (bf16*)alloc((size_t)D * D * 2);
  bf16*  W1b  = (bf16*)alloc((size_t)D * DFF * 2);
  bf16*  W2b  = (bf16*)alloc((size_t)DFF * D * 2);
  bf16*  Qb   = (bf16*)alloc(BSD * 2);
  bf16*  Kb   = (bf16*)alloc(BSD * 2);
  float* attn = (float*)alloc(BSD * 4);
  float* h1   = (float*)alloc(BSD * 4);
  bf16*  h1b  = (bf16*)alloc(BSD * 2);
  bf16*  f1b  = (bf16*)alloc((size_t)M * DFF * 2);
  float* f2   = (float*)alloc(BSD * 4);

  auto cvt = [&](const float* src, bf16* dst, int n) {
    cvt_f32_bf16<<<(n + 255) / 256, 256, 0, stream>>>(src, dst, n);
  };
  cvt(x,  xb,  (int)BSD);
  cvt(Wq, Wqb, D * D);
  cvt(Wk, Wkb, D * D);
  cvt(W1, W1b, D * DFF);
  cvt(W2, W2b, DFF * D);

  // Q = x @ Wq, K = x @ Wk  (bf16 outputs feed attention WMMA directly)
  gemm_wmma_bf16<false, true><<<dim3(D / 128, M / 128), 256, 0, stream>>>(
      xb, Wqb, nullptr, Qb, M, D, D);
  gemm_wmma_bf16<false, true><<<dim3(D / 128, M / 128), 256, 0, stream>>>(
      xb, Wkb, nullptr, Kb, M, D, D);

  // attention (V = x)
  flash_attn<<<dim3(S / 128, H, B), 256, 0, stream>>>(Qb, Kb, xb, pad, attn, S, H, DH);

  // x = LN(x + attn) -> h1 (fp32) + h1b (bf16)
  resid_ln<true><<<M, 256, 0, stream>>>(x, attn, g1, b1, h1, h1b, D);

  // FFN: relu(h1 @ W1) in bf16, then @ W2 in fp32
  gemm_wmma_bf16<true, true><<<dim3(DFF / 128, M / 128), 256, 0, stream>>>(
      h1b, W1b, nullptr, f1b, M, DFF, D);
  gemm_wmma_bf16<false, false><<<dim3(D / 128, M / 128), 256, 0, stream>>>(
      f1b, W2b, f2, nullptr, M, D, DFF);

  // out = LN(h1 + ffn)
  resid_ln<false><<<M, 256, 0, stream>>>(h1, f2, g2, b2, out, nullptr, D);
}
